// SparseBSRLinear_59021440582112
// MI455X (gfx1250) — matrix operand order repaired and checked by reading, not compiled, not run on hardware
//
#include <hip/hip_runtime.h>

// Problem constants (match reference)
#define BATCH   8192
#define IN_F    4096
#define OUT_F   4096
#define BS      64
#define NBR     64
#define BPR     16

// Tiling
#define M_TILE  128
#define THREADS 256
#define XSTR    72   // padded LDS row stride (bf16 elems) = 144 B
#define WSTR    72
#define OSTR    68   // padded f32 stride for output staging (dwords)

typedef __attribute__((ext_vector_type(16))) __bf16 bf16x16;
typedef __attribute__((ext_vector_type(8)))  __bf16 bf16x8;
typedef __attribute__((ext_vector_type(4)))  __bf16 bf16x4;
typedef __attribute__((ext_vector_type(8)))  float  v8f;
typedef __attribute__((ext_vector_type(4)))  float  f32x4;

__device__ __forceinline__ bf16x4 cvt4(f32x4 v) {
    bf16x4 o;
    o.x = (__bf16)v.x; o.y = (__bf16)v.y; o.z = (__bf16)v.z; o.w = (__bf16)v.w;
    return o;
}

__global__ __launch_bounds__(THREADS)
void bsr_linear_wmma(const float* __restrict__ x,
                     const float* __restrict__ vals,
                     const float* __restrict__ bias,
                     const int*   __restrict__ cols,
                     float*       __restrict__ out)
{
    // Xs doubles as the f32 output staging buffer in the epilogue
    // (2*128*72*2 = 36864 B >= 128*68*4 = 34816 B)
    __shared__ __align__(16) __bf16 Xs[2][M_TILE * XSTR];  // 2 x 18 KB
    __shared__ __align__(16) __bf16 Ws[2][BS * WSTR];      // 2 x  9 KB

    const int tid  = threadIdx.x;
    const int lane = tid & 31;
    const int wv   = tid >> 5;       // wave id 0..7
    const int half = lane >> 4;      // 0/1 (half-wave)
    const int l16  = lane & 15;

    // wave tile: 32 rows x 32 cols  (4 M-strips x 2 N-halves)
    const int ms = wv & 3;           // M strip: rows [ms*32, ms*32+32)
    const int nh = wv >> 2;          // N half:  cols [nh*32, nh*32+32)

    const int r  = blockIdx.y;                 // block-row of W
    const int m0 = blockIdx.x * M_TILE;        // batch tile origin

    // staged fp32 tiles (register pipeline)
    f32x4 xr[8];
    f32x4 wr[4];

    // ---- global load of block j into registers (gathered x + W block) ----
    auto load_tiles = [&](int j) {
        const int c = cols[r * BPR + j];       // uniform -> scalar load
        const float* xsrc = x + (size_t)m0 * IN_F + (size_t)c * BS;
        #pragma unroll
        for (int p = 0; p < 8; ++p) {
            int idx = p * THREADS + tid;       // 0..2047
            int row = idx >> 4;                // 0..127
            int col = (idx & 15) << 2;         // 0..60
            xr[p] = *(const f32x4*)(xsrc + (size_t)row * IN_F + col);
        }
        const float* wsrc = vals + (size_t)(r * BPR + j) * (BS * BS);
        #pragma unroll
        for (int p = 0; p < 4; ++p) {
            int idx = p * THREADS + tid;       // 0..1023
            int row = idx >> 4;                // 0..63
            int col = (idx & 15) << 2;
            wr[p] = *(const f32x4*)(wsrc + row * BS + col);
        }
    };

    // ---- convert fp32 -> bf16 and store into LDS buffer ----
    auto store_tiles = [&](int buf) {
        #pragma unroll
        for (int p = 0; p < 8; ++p) {
            int idx = p * THREADS + tid;
            int row = idx >> 4;
            int col = (idx & 15) << 2;
            *(bf16x4*)(&Xs[buf][row * XSTR + col]) = cvt4(xr[p]);
        }
        #pragma unroll
        for (int p = 0; p < 4; ++p) {
            int idx = p * THREADS + tid;
            int row = idx >> 4;
            int col = (idx & 15) << 2;
            *(bf16x4*)(&Ws[buf][row * WSTR + col]) = cvt4(wr[p]);
        }
    };

    // accumulators: acc[mt][nt] = 16x16 tile at rows ms*32+mt*16, cols nh*32+nt*16
    v8f acc[2][2];
    #pragma unroll
    for (int mt = 0; mt < 2; ++mt)
        #pragma unroll
        for (int nt = 0; nt < 2; ++nt)
            #pragma unroll
            for (int e = 0; e < 8; ++e) acc[mt][nt][e] = 0.0f;

    // ---- WMMA compute on LDS buffer ----
    auto compute = [&](int buf) {
        const __bf16* xb = &Xs[buf][0];
        const __bf16* wb = &Ws[buf][0];
        #pragma unroll
        for (int kk = 0; kk < BS; kk += 32) {
            // A fragments 16x32 bf16: elems 0..7 = K kk+8h.., elems 8..15 = K kk+16+8h..
            bf16x16 a[2];
            #pragma unroll
            for (int mt = 0; mt < 2; ++mt) {
                const int arow = ms * 32 + mt * 16 + l16;
                bf16x8 alo = *(const bf16x8*)(xb + arow * XSTR + kk + 8 * half);
                bf16x8 ahi = *(const bf16x8*)(xb + arow * XSTR + kk + 16 + 8 * half);
                #pragma unroll
                for (int i = 0; i < 8; ++i) { a[mt][i] = alo[i]; a[mt][i + 8] = ahi[i]; }
            }
            #pragma unroll
            for (int nt = 0; nt < 2; ++nt) {
                // B fragment 32x16 bf16: lane = N, elems 0..7 = K kk+8h.., elems 8..15 = K kk+16+8h..
                const int n = nh * 32 + nt * 16 + l16;
                bf16x8 blo = *(const bf16x8*)(wb + n * WSTR + kk + 8 * half);
                bf16x8 bhi = *(const bf16x8*)(wb + n * WSTR + kk + 16 + 8 * half);
                bf16x16 b;
                #pragma unroll
                for (int i = 0; i < 8; ++i) { b[i] = blo[i]; b[i + 8] = bhi[i]; }
                #pragma unroll
                for (int mt = 0; mt < 2; ++mt) {
                    acc[mt][nt] = __builtin_amdgcn_wmma_f32_16x16x32_bf16(
                        /*neg_a=*/false, a[mt], /*neg_b=*/false, b,
                        /*c_mod=*/(short)0, acc[mt][nt],
                        /*reuse_a=*/false, /*reuse_b=*/false);
                }
            }
        }
    };

    // ---- software-pipelined main loop over the 16 gathered blocks ----
    load_tiles(0);
    store_tiles(0);
    #pragma unroll 1
    for (int j = 0; j < BPR; ++j) {
        if (j + 1 < BPR) load_tiles(j + 1);        // hide global latency
        __syncthreads();                           // staged buffer j visible
        compute(j & 1);
        if (j + 1 < BPR) store_tiles((j + 1) & 1); // other buffer: no readers now
    }

    // ---- epilogue: stage f32 tile in LDS, then coalesced b128 stores ----
    __syncthreads();                               // all compute reads done; reuse Xs
    float* so = (float*)(&Xs[0][0]);               // 128 x OSTR f32 staging

    // D layout: elem v -> M = v + 8*half, N = l16
    #pragma unroll
    for (int mt = 0; mt < 2; ++mt) {
        #pragma unroll
        for (int nt = 0; nt < 2; ++nt) {
            const int mrow = ms * 32 + mt * 16 + 8 * half;
            const int ncol = nh * 32 + nt * 16 + l16;
            #pragma unroll
            for (int v = 0; v < 8; ++v) {
                so[(mrow + v) * OSTR + ncol] = acc[mt][nt][v];
            }
        }
    }
    __syncthreads();

    // bias for this thread's fixed column group (reused across all 8 passes)
    const int ocol = (tid & 15) << 2;              // 0..60
    f32x4 b4 = *(const f32x4*)(bias + r * BS + ocol);

    #pragma unroll
    for (int p = 0; p < 8; ++p) {
        int idx = p * THREADS + tid;
        int row = idx >> 4;                        // 0..127
        f32x4 v = *(const f32x4*)(so + row * OSTR + ocol);
        v.x += b4.x; v.y += b4.y; v.z += b4.z; v.w += b4.w;
        *(f32x4*)(out + (size_t)(m0 + row) * OUT_F + r * BS + ocol) = v;
    }
}

extern "C" void kernel_launch(void* const* d_in, const int* in_sizes, int n_in,
                              void* d_out, int out_size, void* d_ws, size_t ws_size,
                              hipStream_t stream) {
    const float* x    = (const float*)d_in[0];
    const float* vals = (const float*)d_in[1];
    const float* bias = (const float*)d_in[2];
    const int*   cols = (const int*)d_in[3];
    float* out = (float*)d_out;

    dim3 grid(BATCH / M_TILE, NBR);   // 64 x 64 = 4096 workgroups
    bsr_linear_wmma<<<grid, dim3(THREADS), 0, stream>>>(x, vals, bias, cols, out);
}
